// AudioCausalHead_56925496541349
// MI455X (gfx1250) — matrix operand order repaired and checked by reading, not compiled, not run on hardware
//
#include <hip/hip_runtime.h>
#include <hip/hip_bf16.h>
#include <stdint.h>

// ---------------------------------------------------------------------------
// Types for CDNA5 WMMA
// ---------------------------------------------------------------------------
typedef __attribute__((ext_vector_type(16))) __bf16 v16bf;
typedef __attribute__((ext_vector_type(8)))  __bf16 v8bf;
typedef __attribute__((ext_vector_type(8)))  float  v8f;

static constexpr int Bdim = 16;
static constexpr int Tdim = 2048;
static constexpr int Dm   = 512;
static constexpr size_t BT  = (size_t)Bdim * Tdim;       // 32768
static constexpr size_t BTD = BT * Dm;                   // 16,777,216

// Workspace layout (bytes)
static constexpr size_t OFF_X     = 0;                       // x f32 (64MB), reused as conv out
static constexpr size_t OFF_XR    = OFF_X    + BTD * 4;      // xm -> xr f32 (in place)
static constexpr size_t OFF_XRB   = OFF_XR   + BTD * 4;      // xr bf16 (32MB), reused as yn bf16
static constexpr size_t OFF_V     = OFF_XRB  + BTD * 2;      // v f32 (64MB), reused as y f32
static constexpr size_t OFF_FIELD = OFF_V    + BTD * 4;      // field f32 (8MB)
static constexpr size_t OFF_U     = OFF_FIELD + BT * 64 * 4; // u bf16 (32MB)
static constexpr size_t OFF_SPECB = OFF_U    + BTD * 2;      // spectrogram bf16 padded K=96
static constexpr size_t OFF_WIN   = OFF_SPECB + BT * 96 * 2; // W_in^T bf16 512x96
static constexpr size_t OFF_WV    = OFF_WIN  + 512 * 96 * 2; // W_v^T bf16 512x512
static constexpr size_t OFF_WF    = OFF_WV   + 512 * 512 * 2;// W_field^T bf16 64x512
static constexpr size_t OFF_WO    = OFF_WF   + 64 * 512 * 2; // W_o^T bf16 512x512
static constexpr size_t OFF_WOUT  = OFF_WO   + 512 * 512 * 2;// W_out^T bf16 512x512
static constexpr size_t OFF_FOLD  = OFF_WOUT + 512 * 512 * 2;// 2*512 + 2 floats

__device__ __forceinline__ unsigned short f2bf(float f) {
  unsigned u = __float_as_uint(f);
  unsigned r = u + 0x7FFFu + ((u >> 16) & 1u);
  return (unsigned short)(r >> 16);
}

// ---------------------------------------------------------------------------
// CDNA5 async global->LDS copy (16B per lane), tracked by ASYNCcnt.
// LDS operand: low 32 bits of flat shared address == LDS byte offset.
// ---------------------------------------------------------------------------
__device__ __forceinline__ void async_copy_b128(void* lds_ptr, const void* gptr) {
  uint32_t lds_off = (uint32_t)(uintptr_t)lds_ptr;
  asm volatile("global_load_async_to_lds_b128 %0, %1, off"
               :: "v"(lds_off), "v"((uint64_t)(uintptr_t)gptr)
               : "memory");
}

__device__ __forceinline__ void wait_async0() {
#if __has_builtin(__builtin_amdgcn_s_wait_asynccnt)
  __builtin_amdgcn_s_wait_asynccnt(0);
#else
  asm volatile("s_wait_asynccnt 0x0" ::: "memory");
#endif
}

// ---------------------------------------------------------------------------
// Cast f32 -> bf16 with zero padding (row-major, used for spectrogram)
// dst[r*Cd + c] = (r < Rs && c < Cs) ? bf16(src[r*Cs + c]) : 0
// ---------------------------------------------------------------------------
__global__ void cast_pad_kernel(const float* __restrict__ src,
                                unsigned short* __restrict__ dst,
                                int Rs, int Cs, int Cd, int n) {
  int i = blockIdx.x * 256 + threadIdx.x;
  if (i >= n) return;
  int r = i / Cd, c = i - r * Cd;
  float v = (r < Rs && c < Cs) ? src[(size_t)r * Cs + c] : 0.0f;
  dst[i] = f2bf(v);
}

// ---------------------------------------------------------------------------
// Cast + transpose weights: src (Ks x N row-major f32) -> dst (N x Kd bf16),
// K padded with zeros. Makes GEMM B-tiles contiguous (no in-kernel transpose).
// ---------------------------------------------------------------------------
__global__ void cast_transpose_kernel(const float* __restrict__ src,
                                      unsigned short* __restrict__ dst,
                                      int Ks, int N, int Kd, int n) {
  int i = blockIdx.x * 256 + threadIdx.x;
  if (i >= n) return;
  int row = i / Kd;          // n index
  int k = i - row * Kd;      // k index
  float v = (k < Ks) ? src[(size_t)k * N + row] : 0.0f;
  dst[i] = f2bf(v);
}

// ---------------------------------------------------------------------------
// Fold W_coord (512x8) into even/odd column sums + bias sums
// ---------------------------------------------------------------------------
__global__ void coord_fold_kernel(const float* __restrict__ Wc,
                                  const float* __restrict__ bc,
                                  float* __restrict__ fold) {
  int d = threadIdx.x;  // 512 threads
  float re = 0.f, im = 0.f;
#pragma unroll
  for (int j = 0; j < 4; ++j) {
    re += Wc[d * 8 + 2 * j];
    im += Wc[d * 8 + 2 * j + 1];
  }
  fold[d] = re;
  fold[512 + d] = im;
  if (d == 0) {
    float br = 0.f, bi = 0.f;
#pragma unroll
    for (int j = 0; j < 4; ++j) { br += bc[2 * j]; bi += bc[2 * j + 1]; }
    fold[1024] = br;
    fold[1025] = bi;
  }
}

// ---------------------------------------------------------------------------
// bf16 WMMA GEMM: C(MxN f32) = A(MxK bf16) * Bt^T, Bt is (N x K bf16).
// 256 threads = 8 waves (4 M x 2 N). BK = 32 (one WMMA k-step per stage).
// Double-buffered LDS + GLOBAL_LOAD_ASYNC_TO_LDS_B128 pipeline (ASYNCcnt).
// All dims divisible by tiles -> EXEC all ones for WMMA.
// ---------------------------------------------------------------------------
template <int BM, int BN, bool ADD_RES>
__global__ __launch_bounds__(256) void gemm_bf16_kernel(
    const unsigned short* __restrict__ A, const unsigned short* __restrict__ Bt,
    const float* __restrict__ bias, const float* __restrict__ R,
    float* __restrict__ C, int M, int N, int K) {
  constexpr int BK = 32;
  constexpr int WMr = BM / 4;   // wave tile rows
  constexpr int WNc = BN / 2;   // wave tile cols
  constexpr int MS = WMr / 16;
  constexpr int NS = WNc / 16;
  constexpr int NA = (BM * 4) / 256;  // 16B chunks per thread for A tile
  constexpr int NB = (BN * 4) / 256;  // 16B chunks per thread for B tile

  __shared__ alignas(128) unsigned short As[2][BM][BK];
  __shared__ alignas(128) unsigned short Bs[2][BN][BK];  // [n][k]

  const int tid = threadIdx.x;
  const int m0 = blockIdx.y * BM;
  const int n0 = blockIdx.x * BN;
  const int wid = tid >> 5, lane = tid & 31;
  const int l16 = lane & 15, lh = lane >> 4;
  const int wm = wid & 3, wn = wid >> 2;

  auto load_stage = [&](int k0, int buf) {
#pragma unroll
    for (int j = 0; j < NA; ++j) {
      int c = tid + j * 256;
      int row = c >> 2, kc = (c & 3) * 8;
      async_copy_b128(&As[buf][row][kc],
                      A + (size_t)(m0 + row) * K + k0 + kc);
    }
#pragma unroll
    for (int j = 0; j < NB; ++j) {
      int c = tid + j * 256;
      int row = c >> 2, kc = (c & 3) * 8;
      async_copy_b128(&Bs[buf][row][kc],
                      Bt + (size_t)(n0 + row) * K + k0 + kc);
    }
  };

  v8f acc[MS][NS];
  const v8f vzero = {0.f, 0.f, 0.f, 0.f, 0.f, 0.f, 0.f, 0.f};
#pragma unroll
  for (int i = 0; i < MS; ++i)
#pragma unroll
    for (int j = 0; j < NS; ++j) acc[i][j] = vzero;

  load_stage(0, 0);
  wait_async0();
  __syncthreads();

  const int nstages = K / BK;
  for (int s = 0; s < nstages; ++s) {
    const int buf = s & 1;
    if (s + 1 < nstages) load_stage((s + 1) * BK, buf ^ 1);

    // Fragments use contiguous k = 16*lh + s in both A and B; issue WMMA.
    union AV { v16bf v; v8bf h[2]; };
    v16bf bv[NS];
#pragma unroll
    for (int ns = 0; ns < NS; ++ns) {
      int col = wn * WNc + ns * 16 + l16;
      bv[ns] = *reinterpret_cast<const v16bf*>(&Bs[buf][col][lh * 16]);
    }
#pragma unroll
    for (int ms = 0; ms < MS; ++ms) {
      int row = wm * WMr + ms * 16 + l16;
      AV a;
      a.h[0] = *reinterpret_cast<const v8bf*>(&As[buf][row][lh * 16]);
      a.h[1] = *reinterpret_cast<const v8bf*>(&As[buf][row][lh * 16 + 8]);
#pragma unroll
      for (int ns = 0; ns < NS; ++ns) {
        acc[ms][ns] = __builtin_amdgcn_wmma_f32_16x16x32_bf16(
            false, a.v, false, bv[ns], (short)0, acc[ms][ns], false, false);
      }
    }
    wait_async0();
    __syncthreads();
  }

  // Epilogue: C/D layout — VGPR r holds M=r (lanes 0-15) / M=r+8 (lanes 16-31)
#pragma unroll
  for (int ms = 0; ms < MS; ++ms) {
#pragma unroll
    for (int ns = 0; ns < NS; ++ns) {
      int colg = n0 + wn * WNc + ns * 16 + l16;
#pragma unroll
      for (int r = 0; r < 8; ++r) {
        int rowg = m0 + wm * WMr + ms * 16 + r + 8 * lh;
        float val = acc[ms][ns][r];
        if (bias) val += bias[colg];
        if (ADD_RES) val += R[(size_t)rowg * N + colg];
        C[(size_t)rowg * N + colg] = val;
      }
    }
  }
}

// ---------------------------------------------------------------------------
// Phase + modulation: one wave per token
// ---------------------------------------------------------------------------
__global__ __launch_bounds__(256) void phase_kernel(
    const float* __restrict__ x, const float* __restrict__ fold,
    float* __restrict__ phase_out, float* __restrict__ xm) {
  int token = blockIdx.x * 8 + (threadIdx.x >> 5);
  int lane = threadIdx.x & 31;
  const float* xt = x + (size_t)token * Dm;
  float re = 0.f, im = 0.f;
#pragma unroll
  for (int j = 0; j < 16; ++j) {
    int d = lane + 32 * j;
    float xv = xt[d];
    re += xv * fold[d];
    im += xv * fold[512 + d];
  }
#pragma unroll
  for (int off = 16; off; off >>= 1) {
    re += __shfl_xor(re, off, 32);
    im += __shfl_xor(im, off, 32);
  }
  re += fold[1024];
  im += fold[1025];
  float ph = 0.5f * atan2f(im, re);  // ALPHA = 0.5
  if (lane == 0) phase_out[token] = ph;
  float cp = cosf(ph);
#pragma unroll
  for (int j = 0; j < 16; ++j) {
    int d = lane + 32 * j;
    xm[(size_t)token * Dm + d] = xt[d] * cp;
  }
}

// ---------------------------------------------------------------------------
// LiorKernel recurrence: one thread per (b,d), sequential over T.
// ---------------------------------------------------------------------------
__global__ __launch_bounds__(256) void lior_kernel(
    float* __restrict__ xr, unsigned short* __restrict__ xrb,
    const float* __restrict__ tau_exp, const float* __restrict__ tau_frac,
    const float* __restrict__ tau_osc) {
  int idx = blockIdx.x * 256 + threadIdx.x;  // 0..8191
  int b = idx >> 9, d = idx & 511;
  float te = tau_exp[0], tf = tau_frac[0], to = tau_osc[0];
  float decay = expf(-1.0f / te);
  float c[4];
#pragma unroll
  for (int k = 0; k < 4; ++k)
    c[k] = cosf((float)k / to) * powf((float)k + 1.0f, -1.0f / tf);
  float m = 0.f, h0 = 0.f, h1 = 0.f, h2 = 0.f, h3 = 0.f;
  const size_t base = ((size_t)b * Tdim) * Dm + d;
  for (int t = 0; t < Tdim; ++t) {
    size_t p = base + (size_t)t * Dm;
    float xt = xr[p];
    m = decay * m + (1.0f - decay) * xt + h0 * c[0] + h1 * c[1] + h2 * c[2] + h3 * c[3];
    h3 = h2; h2 = h1; h1 = h0; h0 = xt;
    float y = xt + m;
    xr[p] = y;
    xrb[p] = f2bf(y);
  }
}

// ---------------------------------------------------------------------------
// Causal depthwise conv, K=64 taps. Tile: 64 t x 64 d with 63-row halo in LDS.
// ---------------------------------------------------------------------------
__global__ __launch_bounds__(256) void dwconv_kernel(
    const float* __restrict__ v, const float* __restrict__ w,
    float* __restrict__ out) {
  __shared__ float vt[127][64];
  __shared__ float wt[64][64];
  int d0 = blockIdx.x * 64;
  int t0 = blockIdx.y * 64;
  int b = blockIdx.z;
  int tid = threadIdx.x;
  const size_t vbase = ((size_t)b * Tdim) * Dm;
  for (int i = tid; i < 127 * 64; i += 256) {
    int r = i >> 6, c = i & 63;
    int t = t0 - 63 + r;
    vt[r][c] = (t >= 0) ? v[vbase + (size_t)t * Dm + d0 + c] : 0.f;
  }
  for (int i = tid; i < 64 * 64; i += 256) {
    int k = i & 63, dd = i >> 6;
    wt[k][dd] = w[(size_t)(d0 + dd) * 64 + k];
  }
  __syncthreads();
  int tx = tid & 63, ty = tid >> 6;
  for (int tt = ty; tt < 64; tt += 4) {
    float acc = 0.f;
#pragma unroll
    for (int k = 0; k < 64; ++k) acc += vt[tt + k][tx] * wt[k][tx];
    out[vbase + (size_t)(t0 + tt) * Dm + d0 + tx] = acc;
  }
}

// ---------------------------------------------------------------------------
// Spinor gate * SiLU(conv): one block per token, W_gate cached in LDS.
// ---------------------------------------------------------------------------
__global__ __launch_bounds__(256) void gate_kernel(
    const float* __restrict__ field, const float* __restrict__ Wg,
    const float* __restrict__ conv, unsigned short* __restrict__ u) {
  __shared__ float q[16];
  __shared__ float wg_s[16 * 512];
  int token = blockIdx.x;
  int tid = threadIdx.x;
  for (int i = tid; i < 16 * 512; i += 256) wg_s[i] = Wg[i];
  if (tid < 16) {
    const float* f = field + (size_t)token * 64 + tid * 4;
    q[tid] = f[0] * f[0] + f[1] * f[1] + f[2] * f[2] + f[3] * f[3];
  }
  __syncthreads();
  for (int d = tid; d < 512; d += 256) {
    float g = 0.f;
#pragma unroll
    for (int f = 0; f < 16; ++f) g += q[f] * wg_s[f * 512 + d];
    g = 1.f / (1.f + expf(-g));                 // sigmoid
    float cv = conv[(size_t)token * Dm + d];
    float s = cv / (1.f + expf(-cv));           // silu
    u[(size_t)token * Dm + d] = f2bf(s * g);
  }
}

// ---------------------------------------------------------------------------
// LayerNorm -> bf16. One block of 256 per token (2 elems/thread).
// ---------------------------------------------------------------------------
__global__ __launch_bounds__(256) void ln_kernel(
    const float* __restrict__ y, const float* __restrict__ g,
    const float* __restrict__ b, unsigned short* __restrict__ yn) {
  int token = blockIdx.x;
  int tid = threadIdx.x;
  const float* yt = y + (size_t)token * Dm;
  float v0 = yt[tid], v1 = yt[tid + 256];
  float s = v0 + v1, ss = v0 * v0 + v1 * v1;
#pragma unroll
  for (int off = 16; off; off >>= 1) {
    s += __shfl_xor(s, off, 32);
    ss += __shfl_xor(ss, off, 32);
  }
  __shared__ float sb[8], ssb[8];
  __shared__ float mu_s, rstd_s;
  int w = tid >> 5;
  if ((tid & 31) == 0) { sb[w] = s; ssb[w] = ss; }
  __syncthreads();
  if (tid == 0) {
    float S = 0.f, SS = 0.f;
#pragma unroll
    for (int i = 0; i < 8; ++i) { S += sb[i]; SS += ssb[i]; }
    float mu = S / 512.0f;
    float var = SS / 512.0f - mu * mu;
    mu_s = mu;
    rstd_s = rsqrtf(var + 1e-5f);
  }
  __syncthreads();
  float mu = mu_s, r = rstd_s;
  yn[(size_t)token * Dm + tid]       = f2bf((v0 - mu) * r * g[tid] + b[tid]);
  yn[(size_t)token * Dm + tid + 256] = f2bf((v1 - mu) * r * g[tid + 256] + b[tid + 256]);
}

// ---------------------------------------------------------------------------
// Host-side launch
// ---------------------------------------------------------------------------
extern "C" void kernel_launch(void* const* d_in, const int* in_sizes, int n_in,
                              void* d_out, int out_size, void* d_ws, size_t ws_size,
                              hipStream_t stream) {
  (void)in_sizes; (void)n_in; (void)out_size; (void)ws_size;
  const float* spec    = (const float*)d_in[0];
  const float* W_in    = (const float*)d_in[1];
  const float* b_in    = (const float*)d_in[2];
  const float* W_coord = (const float*)d_in[3];
  const float* b_coord = (const float*)d_in[4];
  const float* tau_exp = (const float*)d_in[5];
  const float* tau_frac= (const float*)d_in[6];
  const float* tau_osc = (const float*)d_in[7];
  const float* W_v     = (const float*)d_in[8];
  const float* conv_w  = (const float*)d_in[9];
  const float* W_field = (const float*)d_in[10];
  const float* W_gate  = (const float*)d_in[11];
  const float* W_o     = (const float*)d_in[12];
  const float* ln_g    = (const float*)d_in[13];
  const float* ln_b    = (const float*)d_in[14];
  const float* W_out   = (const float*)d_in[15];
  const float* b_out   = (const float*)d_in[16];

  float* out = (float*)d_out;
  float* phase_out = out + BTD;

  char* ws = (char*)d_ws;
  float* X     = (float*)(ws + OFF_X);
  float* CONV  = (float*)(ws + OFF_X);      // reuse: x dead after phase kernel
  float* XR    = (float*)(ws + OFF_XR);
  unsigned short* XRB  = (unsigned short*)(ws + OFF_XRB);
  unsigned short* YNB  = (unsigned short*)(ws + OFF_XRB);  // reuse after W_o GEMM
  float* V     = (float*)(ws + OFF_V);
  float* Y     = (float*)(ws + OFF_V);      // reuse: v dead after conv
  float* FIELD = (float*)(ws + OFF_FIELD);
  unsigned short* U    = (unsigned short*)(ws + OFF_U);
  unsigned short* SPECB= (unsigned short*)(ws + OFF_SPECB);
  unsigned short* WINB = (unsigned short*)(ws + OFF_WIN);
  unsigned short* WVB  = (unsigned short*)(ws + OFF_WV);
  unsigned short* WFB  = (unsigned short*)(ws + OFF_WF);
  unsigned short* WOB  = (unsigned short*)(ws + OFF_WO);
  unsigned short* WOUTB= (unsigned short*)(ws + OFF_WOUT);
  float* FOLD  = (float*)(ws + OFF_FOLD);

  auto cdiv = [](int a, int b) { return (a + b - 1) / b; };

  // 1) casts: spectrogram padded row-major; weights transposed to N x K bf16
  { int n = (int)BT * 96;
    cast_pad_kernel<<<cdiv(n, 256), 256, 0, stream>>>(spec, SPECB, (int)BT, 80, 96, n); }
  { int n = 512 * 96;   // W_in (80x512) -> (512x96)
    cast_transpose_kernel<<<cdiv(n, 256), 256, 0, stream>>>(W_in, WINB, 80, 512, 96, n); }
  { int n = 512 * 512;  // W_v -> W_v^T
    cast_transpose_kernel<<<cdiv(n, 256), 256, 0, stream>>>(W_v, WVB, 512, 512, 512, n); }
  { int n = 64 * 512;   // W_field (512x64) -> (64x512)
    cast_transpose_kernel<<<cdiv(n, 256), 256, 0, stream>>>(W_field, WFB, 512, 64, 512, n); }
  { int n = 512 * 512;  // W_o -> W_o^T
    cast_transpose_kernel<<<cdiv(n, 256), 256, 0, stream>>>(W_o, WOB, 512, 512, 512, n); }
  { int n = 512 * 512;  // W_out -> W_out^T
    cast_transpose_kernel<<<cdiv(n, 256), 256, 0, stream>>>(W_out, WOUTB, 512, 512, 512, n); }

  // 2) fold coord projection
  coord_fold_kernel<<<1, 512, 0, stream>>>(W_coord, b_coord, FOLD);

  // 3) input projection GEMM: x = spec96 @ Win + b_in
  gemm_bf16_kernel<128, 128, false><<<dim3(4, (int)BT / 128), 256, 0, stream>>>(
      SPECB, WINB, b_in, nullptr, X, (int)BT, 512, 96);

  // 4) phase + modulation
  phase_kernel<<<(int)BT / 8, 256, 0, stream>>>(X, FOLD, phase_out, XR);

  // 5) recurrence (in-place xm -> xr, plus bf16 copy)
  lior_kernel<<<(Bdim * Dm) / 256, 256, 0, stream>>>(XR, XRB, tau_exp, tau_frac, tau_osc);

  // 6) v = xr @ W_v
  gemm_bf16_kernel<128, 128, false><<<dim3(4, (int)BT / 128), 256, 0, stream>>>(
      XRB, WVB, nullptr, nullptr, V, (int)BT, 512, 512);

  // 7) field = xr @ W_field (N = 64)
  gemm_bf16_kernel<128, 64, false><<<dim3(1, (int)BT / 128), 256, 0, stream>>>(
      XRB, WFB, nullptr, nullptr, FIELD, (int)BT, 64, 512);

  // 8) causal depthwise conv
  dwconv_kernel<<<dim3(Dm / 64, Tdim / 64, Bdim), 256, 0, stream>>>(V, conv_w, CONV);

  // 9) u = silu(conv) * sigmoid((sum field^2) @ W_gate)
  gate_kernel<<<(int)BT, 256, 0, stream>>>(FIELD, W_gate, CONV, U);

  // 10) y = u @ W_o + xr (residual)
  gemm_bf16_kernel<128, 128, true><<<dim3(4, (int)BT / 128), 256, 0, stream>>>(
      U, WOB, nullptr, XR, Y, (int)BT, 512, 512);

  // 11) LayerNorm -> bf16
  ln_kernel<<<(int)BT, 256, 0, stream>>>(Y, ln_g, ln_b, YNB);

  // 12) out = yn @ W_out + b_out
  gemm_bf16_kernel<128, 128, false><<<dim3(4, (int)BT / 128), 256, 0, stream>>>(
      YNB, WOUTB, b_out, nullptr, out, (int)BT, 512, 512);
}
